// MemoryEnhancedGating_14516989460793
// MI455X (gfx1250) — matrix operand order repaired and checked by reading, not compiled, not run on hardware
//
#include <hip/hip_runtime.h>
#include <hip/hip_bf16.h>

typedef __attribute__((ext_vector_type(16))) _Float16 v16h;
typedef __attribute__((ext_vector_type(8)))  _Float16 v8h;
typedef __attribute__((ext_vector_type(8)))  float    v8f;

#define T_TOKENS 1048576
#define TM 128          // tokens per GEMM tile / workgroup
#define SEG 4096        // tokens per scan segment
#define NSEG (T_TOKENS / SEG)   // 256

// ---------------- CDNA5 async global->LDS copy (ASYNCcnt path) ----------------
// Each lane copies 16 bytes from its global address straight into LDS at its
// LDS byte offset (no VGPR round trip). Tracked by ASYNCcnt.
__device__ __forceinline__ void async_copy_b128(unsigned lds_off, const void* gptr) {
  asm volatile("global_load_async_to_lds_b128 %0, %1, off"
               :: "v"(lds_off), "v"(gptr)
               : "memory");
}
__device__ __forceinline__ void wait_async() {
  asm volatile("s_wait_asynccnt 0" ::: "memory");
}
// Generic shared-pointer -> LDS byte offset (low 32 bits of the aperture addr).
__device__ __forceinline__ unsigned lds_offset(const void* p) {
  return (unsigned)(unsigned long long)p;
}

// ---------------- WMMA fragment loaders (LDS, f16) ----------------
// A (16x32, MxK): lane l -> m = l&15, g = l>>4.
//   half h in 0..7  : K = k0 + g*8 + h
//   half h in 8..15 : K = k0 + 16 + g*8 + (h-8)
__device__ __forceinline__ v16h load_frag_a(const _Float16* base, int rowlen,
                                            int m, int g, int k0) {
  const _Float16* r = base + m * rowlen + k0 + g * 8;
  v8h lo = *(const v8h*)(r);
  v8h hi = *(const v8h*)(r + 16);
  v16h a;
#pragma unroll
  for (int i = 0; i < 8; ++i) { a[i] = lo[i]; a[i + 8] = hi[i]; }
  return a;
}

// B (32x16, KxN) from a transposed weight tile WT[n][k]:
//   lane l -> n = l&15, g = l>>4; half h in 0..7 -> K = k0 + g*8 + h, 8..15 -> K = k0 + 16 + g*8 + (h-8)
__device__ __forceinline__ v16h load_frag_b(const _Float16* baseT, int rowlen,
                                            int n, int g, int k0) {
  const _Float16* r = baseT + n * rowlen + k0 + g * 8;
  v8h lo = *(const v8h*)(r);
  v8h hi = *(const v8h*)(r + 16);
  v16h b;
#pragma unroll
  for (int i = 0; i < 8; ++i) { b[i] = lo[i]; b[i + 8] = hi[i]; }
  return b;
}

__device__ __forceinline__ v8f wmma_f16(v16h a, v16h b, v8f c) {
  return __builtin_amdgcn_wmma_f32_16x16x32_f16(false, a, false, b, (short)0, c,
                                                false, false);
}

// ---------------- Kernel 1: expert 0 over all tokens ----------------
__global__ __launch_bounds__(256) void k_expert0(
    const float* __restrict__ x, const int* __restrict__ topk,
    const float* __restrict__ W0a, const float* __restrict__ b0a,
    const float* __restrict__ W0b, const float* __restrict__ b0b,
    float* __restrict__ out0) {
  __shared__ __align__(16) float    Xs32[TM * 64];     // async-staged f32 x tile
  __shared__ __align__(16) _Float16 Xs[TM * 64];       // X tile, f16 row-major
  __shared__ __align__(16) _Float16 WaT[128 * 64];     // W0a^T: [h][d]
  __shared__ __align__(16) _Float16 WbT[32 * 128];     // W0b^T: [o][h]
  __shared__ __align__(16) _Float16 Hs[TM * 128];      // hidden, row-major
  __shared__ float ba[128];
  __shared__ float bb[32];
  __shared__ int   m0s[TM];

  const int tid = threadIdx.x;
  const long long tok0 = (long long)blockIdx.x * TM;

  // Kick off the async HBM->LDS stream of the x tile (contiguous 32 KB).
  {
    const float* gsrc = x + tok0 * 64;
    unsigned base_off = lds_offset(Xs32);
#pragma unroll
    for (int i = 0; i < (TM * 64) / (256 * 4); ++i) {  // 8 iters of b128/lane
      int e4 = i * 256 + tid;                          // float4 index
      async_copy_b128(base_off + e4 * 16, gsrc + e4 * 4);
    }
  }

  // Overlap: transpose + f16-convert weights while x streams in.
  for (int i = tid; i < 64 * 128; i += 256) {          // W0a is [64][128]
    int d = i >> 7, h = i & 127;
    WaT[h * 64 + d] = (_Float16)W0a[i];
  }
  for (int i = tid; i < 128 * 32; i += 256) {          // W0b is [128][32]
    int k = i >> 5, n = i & 31;
    WbT[n * 128 + k] = (_Float16)W0b[i];
  }
  if (tid < 128) ba[tid] = b0a[tid];
  if (tid < 32)  bb[tid] = b0b[tid];
  if (tid < TM) {
    long long gt = tok0 + tid;
    int a = topk[2 * gt], b = topk[2 * gt + 1];
    m0s[tid] = (a == 0) || (b == 0);
  }
  wait_async();
  __syncthreads();
  // Convert staged f32 tile -> f16 tile.
  for (int i = tid; i < TM * 64; i += 256) Xs[i] = (_Float16)Xs32[i];
  __syncthreads();

  const int wave = tid >> 5, lane = tid & 31;
  const int m = lane & 15, g = lane >> 4;

  // GEMM1: Hs = relu(Xs @ W0a + b0a)   (128x64 @ 64x128)
  {
    const int mb = wave;  // 8 waves cover 8 M-blocks
    v16h a0 = load_frag_a(Xs, 64, mb * 16 + m, g, 0);
    v16h a1 = load_frag_a(Xs, 64, mb * 16 + m, g, 32);
#pragma unroll
    for (int nb = 0; nb < 8; ++nb) {
      v8f acc = {};
      acc = wmma_f16(a0, load_frag_b(WaT, 64, nb * 16 + m, g, 0), acc);
      acc = wmma_f16(a1, load_frag_b(WaT, 64, nb * 16 + m, g, 32), acc);
      float bias = ba[nb * 16 + m];
#pragma unroll
      for (int r = 0; r < 8; ++r) {
        float v = acc[r] + bias;
        v = v > 0.0f ? v : 0.0f;
        Hs[(mb * 16 + g * 8 + r) * 128 + nb * 16 + m] = (_Float16)v;
      }
    }
  }
  __syncthreads();

  // GEMM2: e0 = Hs @ W0b + b0b         (128x128 @ 128x32)
  {
    const int mb = wave;
    v16h af[4];
#pragma unroll
    for (int kk = 0; kk < 4; ++kk)
      af[kk] = load_frag_a(Hs, 128, mb * 16 + m, g, kk * 32);
#pragma unroll
    for (int nb = 0; nb < 2; ++nb) {
      v8f acc = {};
#pragma unroll
      for (int kk = 0; kk < 4; ++kk)
        acc = wmma_f16(af[kk], load_frag_b(WbT, 128, nb * 16 + m, g, kk * 32), acc);
      float bias = bb[nb * 16 + m];
#pragma unroll
      for (int r = 0; r < 8; ++r) {
        int t = mb * 16 + g * 8 + r;
        long long gt = tok0 + t;
        float v = acc[r] + bias;
        out0[gt * 32 + nb * 16 + m] = m0s[t] ? v : 0.0f;
      }
    }
  }
}

// ---------------- Scan kernels: idx = cummax(where(mask0, t, 0)) ----------------
__global__ __launch_bounds__(256) void k_seg_last(const int* __restrict__ topk,
                                                  int* __restrict__ segLast) {
  __shared__ int s[256];
  const int tid = threadIdx.x;
  const long long base = (long long)blockIdx.x * SEG;
  int best = 0;
  for (int i = tid; i < SEG; i += 256) {
    long long t = base + i;
    int a = topk[2 * t], b = topk[2 * t + 1];
    if (a == 0 || b == 0) best = (best > (int)t) ? best : (int)t;
  }
  s[tid] = best;
  __syncthreads();
#pragma unroll
  for (int off = 128; off > 0; off >>= 1) {
    if (tid < off) s[tid] = max(s[tid], s[tid + off]);
    __syncthreads();
  }
  if (tid == 0) segLast[blockIdx.x] = s[0];
}

__global__ __launch_bounds__(256) void k_seg_scan(const int* __restrict__ segLast,
                                                  int* __restrict__ segCarry) {
  __shared__ int s[256];
  const int tid = threadIdx.x;
  s[tid] = segLast[tid];
  __syncthreads();
#pragma unroll
  for (int off = 1; off < 256; off <<= 1) {
    int v = (tid >= off) ? s[tid - off] : 0;
    __syncthreads();
    if (tid >= off) s[tid] = max(s[tid], v);
    __syncthreads();
  }
  segCarry[tid] = (tid > 0) ? s[tid - 1] : 0;  // exclusive max
}

__global__ __launch_bounds__(256) void k_fill_idx(const int* __restrict__ topk,
                                                  const int* __restrict__ segCarry,
                                                  int* __restrict__ idx) {
  __shared__ int s[256];
  const int tid = threadIdx.x;
  const long long base = (long long)blockIdx.x * SEG;
  const long long t0 = base + (long long)tid * 16;
  int last = 0;
#pragma unroll
  for (int i = 0; i < 16; ++i) {
    long long t = t0 + i;
    int a = topk[2 * t], b = topk[2 * t + 1];
    if (a == 0 || b == 0) last = (int)t;
  }
  s[tid] = last;
  __syncthreads();
#pragma unroll
  for (int off = 1; off < 256; off <<= 1) {
    int v = (tid >= off) ? s[tid - off] : 0;
    __syncthreads();
    if (tid >= off) s[tid] = max(s[tid], v);
    __syncthreads();
  }
  int run = segCarry[blockIdx.x];
  if (tid > 0) run = max(run, s[tid - 1]);
#pragma unroll
  for (int i = 0; i < 16; ++i) {
    long long t = t0 + i;
    int a = topk[2 * t], b = topk[2 * t + 1];
    if (a == 0 || b == 0) run = (int)t;
    idx[t] = run;
  }
}

// ---------------- Kernel 5: expert 1 + final blend ----------------
__global__ __launch_bounds__(256) void k_expert1(
    const float* __restrict__ x, const int* __restrict__ topk,
    const float* __restrict__ wgt,
    const float* __restrict__ W1a, const float* __restrict__ b1a,
    const float* __restrict__ W1b, const float* __restrict__ b1b,
    const float* __restrict__ out0, const int* __restrict__ idx,
    float* __restrict__ out) {
  __shared__ __align__(16) float    Xs32[TM * 64];     // async-staged f32 x tile
  __shared__ __align__(16) _Float16 Xs[TM * 96];       // [x | filled], f16
  __shared__ __align__(16) _Float16 WaT[128 * 96];     // W1a^T: [h][k]
  __shared__ __align__(16) _Float16 WbT[32 * 128];     // W1b^T: [o][h]
  __shared__ __align__(16) _Float16 Hs[TM * 128];
  __shared__ float ba[128];
  __shared__ float bb[32];
  __shared__ float wv[TM];
  __shared__ int   m1s[TM];

  const int tid = threadIdx.x;
  const long long tok0 = (long long)blockIdx.x * TM;

  // Async-stage the x part of the tile.
  {
    const float* gsrc = x + tok0 * 64;
    unsigned base_off = lds_offset(Xs32);
#pragma unroll
    for (int i = 0; i < (TM * 64) / (256 * 4); ++i) {
      int e4 = i * 256 + tid;
      async_copy_b128(base_off + e4 * 16, gsrc + e4 * 4);
    }
  }

  // Overlap: weights transpose + gather of filled rows (vectorized b128 loads).
  for (int i = tid; i < 96 * 128; i += 256) {          // W1a is [96][128]
    int k = i >> 7, h = i & 127;
    WaT[h * 96 + k] = (_Float16)W1a[i];
  }
  for (int i = tid; i < 128 * 32; i += 256) {          // W1b is [128][32]
    int k = i >> 5, n = i & 31;
    WbT[n * 128 + k] = (_Float16)W1b[i];
  }
  if (tid < 128) ba[tid] = b1a[tid];
  if (tid < 32)  bb[tid] = b1b[tid];
  {
    const float4* o0v = (const float4*)out0;           // 8 float4 per row
    for (int i = tid; i < TM * 8; i += 256) {
      int t = i >> 3, c4 = i & 7;
      long long gt = tok0 + t;
      long long src = (long long)idx[gt];
      float4 v = o0v[src * 8 + c4];
      int b = t * 96 + 64 + c4 * 4;
      Xs[b + 0] = (_Float16)v.x;
      Xs[b + 1] = (_Float16)v.y;
      Xs[b + 2] = (_Float16)v.z;
      Xs[b + 3] = (_Float16)v.w;
    }
  }
  if (tid < TM) {
    long long gt = tok0 + tid;
    int a = topk[2 * gt], b = topk[2 * gt + 1];
    m1s[tid] = (a == 1) || (b == 1);
    wv[tid] = wgt[2 * gt];
  }
  wait_async();
  __syncthreads();
  for (int i = tid; i < TM * 64; i += 256) {
    int t = i >> 6, d = i & 63;
    Xs[t * 96 + d] = (_Float16)Xs32[i];
  }
  __syncthreads();

  const int wave = tid >> 5, lane = tid & 31;
  const int m = lane & 15, g = lane >> 4;

  // GEMM1: Hs = relu(X1 @ W1a + b1a)   (128x96 @ 96x128)
  {
    const int mb = wave;
    v16h a0 = load_frag_a(Xs, 96, mb * 16 + m, g, 0);
    v16h a1 = load_frag_a(Xs, 96, mb * 16 + m, g, 32);
    v16h a2 = load_frag_a(Xs, 96, mb * 16 + m, g, 64);
#pragma unroll
    for (int nb = 0; nb < 8; ++nb) {
      v8f acc = {};
      acc = wmma_f16(a0, load_frag_b(WaT, 96, nb * 16 + m, g, 0), acc);
      acc = wmma_f16(a1, load_frag_b(WaT, 96, nb * 16 + m, g, 32), acc);
      acc = wmma_f16(a2, load_frag_b(WaT, 96, nb * 16 + m, g, 64), acc);
      float bias = ba[nb * 16 + m];
#pragma unroll
      for (int r = 0; r < 8; ++r) {
        float v = acc[r] + bias;
        v = v > 0.0f ? v : 0.0f;
        Hs[(mb * 16 + g * 8 + r) * 128 + nb * 16 + m] = (_Float16)v;
      }
    }
  }
  __syncthreads();

  // GEMM2 + blend: out = w*out0 + (1-w)*(mask1 ? e1 : 0)
  {
    const int mb = wave;
    v16h af[4];
#pragma unroll
    for (int kk = 0; kk < 4; ++kk)
      af[kk] = load_frag_a(Hs, 128, mb * 16 + m, g, kk * 32);
#pragma unroll
    for (int nb = 0; nb < 2; ++nb) {
      v8f acc = {};
#pragma unroll
      for (int kk = 0; kk < 4; ++kk)
        acc = wmma_f16(af[kk], load_frag_b(WbT, 128, nb * 16 + m, g, kk * 32), acc);
      float bias = bb[nb * 16 + m];
#pragma unroll
      for (int r = 0; r < 8; ++r) {
        int t = mb * 16 + g * 8 + r;
        long long gt = tok0 + t;
        int col = nb * 16 + m;
        float e1 = acc[r] + bias;
        float o1 = m1s[t] ? e1 : 0.0f;
        float o0 = out0[gt * 32 + col];
        float w = wv[t];
        out[gt * 32 + col] = w * o0 + (1.0f - w) * o1;
      }
    }
  }
}

extern "C" void kernel_launch(void* const* d_in, const int* in_sizes, int n_in,
                              void* d_out, int out_size, void* d_ws, size_t ws_size,
                              hipStream_t stream) {
  const float* x    = (const float*)d_in[0];
  const int*   topk = (const int*)d_in[1];
  const float* wgt  = (const float*)d_in[2];
  const float* W0a  = (const float*)d_in[3];
  const float* b0a  = (const float*)d_in[4];
  const float* W0b  = (const float*)d_in[5];
  const float* b0b  = (const float*)d_in[6];
  const float* W1a  = (const float*)d_in[7];
  const float* b1a  = (const float*)d_in[8];
  const float* W1b  = (const float*)d_in[9];
  const float* b1b  = (const float*)d_in[10];
  float* out = (float*)d_out;

  char* ws = (char*)d_ws;
  float* out0     = (float*)ws;                                   // T*32 f32
  int*   idxArr   = (int*)(ws + (size_t)T_TOKENS * 32 * 4);       // T i32
  int*   segLast  = (int*)(ws + (size_t)T_TOKENS * 32 * 4 + (size_t)T_TOKENS * 4);
  int*   segCarry = segLast + NSEG;

  dim3 blk(256);
  k_expert0<<<T_TOKENS / TM, blk, 0, stream>>>(x, topk, W0a, b0a, W0b, b0b, out0);
  k_seg_last<<<NSEG, blk, 0, stream>>>(topk, segLast);
  k_seg_scan<<<1, blk, 0, stream>>>(segLast, segCarry);
  k_fill_idx<<<NSEG, blk, 0, stream>>>(topk, segCarry, idxArr);
  k_expert1<<<T_TOKENS / TM, blk, 0, stream>>>(x, topk, wgt, W1a, b1a, W1b, b1b,
                                               out0, idxArr, out);
}
